// ExactSpectralHead_1382979469665
// MI455X (gfx1250) — matrix operand order repaired and checked by reading, not compiled, not run on hardware
//
#include <hip/hip_runtime.h>
#include <hip/hip_bf16.h>

typedef __attribute__((ext_vector_type(16))) _Float16 v16h;
typedef __attribute__((ext_vector_type(8)))  float    v8f;
typedef __attribute__((ext_vector_type(4)))  unsigned int u32x4;
typedef __attribute__((ext_vector_type(8)))  int          i32x8;
typedef __attribute__((ext_vector_type(4)))  int          i32x4;

#define WMMA_F16(A, B, C) \
  __builtin_amdgcn_wmma_f32_16x16x32_f16(false, (A), false, (B), (short)0, (C), false, false)

// Problem constants (fixed by the reference): B=8, T=2048, C=1024, H=128
#define DIM_B 8
#define DIM_T 2048
#define DIM_C 1024
#define DIM_H 128
#define SCALE 0.03125f   // C^-0.5 = 1/32
#define NEGHUGE (-1.0e30f)

#if defined(__gfx1250__) && __has_builtin(__builtin_amdgcn_tensor_load_to_lds)
#define HAVE_TDM 1
#else
#define HAVE_TDM 0
#endif

// ---------------------------------------------------------------------------
// CDNA5 16-bit A-matrix 16x32 layout (ISA 7.12.2)
// ---------------------------------------------------------------------------
__device__ __forceinline__ int a_k_of(int e, int lane) {
  int g = e >> 1, h = e & 1;
  int k = (g < 4) ? (2 * g) : (16 + 2 * (g - 4));
  return k + ((lane >> 4) << 3) + h;
}
__device__ __forceinline__ int b_k_of(int e, int lane) {
  int g = e >> 1, h = e & 1;
  int k = (g < 4) ? (2 * g) : (16 + 2 * (g - 4));
  return k + ((lane >> 4) << 3) + h;
}

__device__ __forceinline__ v16h load_a_f16(const _Float16* src, int ld, int lane) {
  v16h a;
  int m = lane & 15;
#pragma unroll
  for (int e = 0; e < 16; ++e) a[e] = src[(long)m * ld + a_k_of(e, lane)];
  return a;
}
// B tile: element (kk,n) at src[kk*strideK + n*strideN]; strideK==1 -> b128 loads
__device__ __forceinline__ v16h load_b_f16(const _Float16* src, long strideK, long strideN, int lane) {
  v16h b;
  int n = lane & 15;
#pragma unroll
  for (int e = 0; e < 16; ++e) b[e] = src[(long)b_k_of(e, lane) * strideK + (long)n * strideN];
  return b;
}
__device__ __forceinline__ v16h load_b_f32cvt(const float* src, long strideK, long strideN, int lane) {
  v16h b;
  int n = lane & 15;
#pragma unroll
  for (int e = 0; e < 16; ++e)
    b[e] = (_Float16)src[(long)b_k_of(e, lane) * strideK + (long)n * strideN];
  return b;
}

__device__ __forceinline__ float rowmax16(float v) {
#pragma unroll
  for (int m = 1; m < 16; m <<= 1) v = fmaxf(v, __shfl_xor(v, m, 32));
  return v;
}
__device__ __forceinline__ float rowsum16(float v) {
#pragma unroll
  for (int m = 1; m < 16; m <<= 1) v += __shfl_xor(v, m, 32);
  return v;
}
__device__ __forceinline__ void wait_ds0() {
#if __has_builtin(__builtin_amdgcn_s_wait_dscnt)
  __builtin_amdgcn_s_wait_dscnt(0);
#else
  asm volatile("s_wait_dscnt 0" ::: "memory");
#endif
}

#if HAVE_TDM
// Issue a 2-D TDM tile load (f16 elements) into LDS.  D# layout per ISA §8.3/8.4:
// group0: count=1 | lds_addr | global_addr[56:0] | type=2
// group1: data_size=1(2B) | tensor_dim0/1 | tile_dim0/1 | tensor_dim0_stride
// This toolchain's builtin takes 6 args (g0, g1, g2, g3, extra group, cpol).
__device__ __forceinline__ void tdm_load_2d_f16(unsigned lds_addr, const void* gptr,
                                                unsigned tensor_d0, unsigned tensor_d1,
                                                unsigned tile_d0, unsigned tile_d1,
                                                unsigned stride0) {
  unsigned long long ga = (unsigned long long)(size_t)gptr;
  u32x4 g0;
  g0[0] = 1u;                                               // count=1, user descriptor
  g0[1] = lds_addr;                                         // bits 63:32
  g0[2] = (unsigned)(ga & 0xffffffffu);                     // bits 95:64
  g0[3] = (unsigned)((ga >> 32) & 0x01ffffffu) | (2u << 30);// addr[56:32] | type=2
  i32x8 g1;
  g1[0] = (int)(1u << 16);                                  // data_size=1 -> 2 bytes
  g1[1] = (int)(tensor_d0 << 16);                           // tensor_dim0[15:0]
  g1[2] = (int)((tensor_d0 >> 16) | (tensor_d1 << 16));     // dim0 hi | dim1 lo
  g1[3] = (int)((tensor_d1 >> 16) | (tile_d0 << 16));       // dim1 hi | tile_dim0
  g1[4] = (int)(tile_d1 & 0xffffu);                         // tile_dim1, tile_dim2=0
  g1[5] = (int)stride0;                                     // tensor_dim0_stride lo
  g1[6] = 0;
  g1[7] = 0;
  i32x4 z4 = {};
  i32x8 z8 = {};
  __builtin_amdgcn_tensor_load_to_lds(g0, g1, z4, z4, z8, 0);
}
#endif

// ---------------------------------------------------------------------------
// Kernel 1: QKV projection.  grid = B*T/16 blocks, 128 threads (4 waves).
// q,k stored row-major [B*T][H] f16; v stored TRANSPOSED [B][H][T] f16 so the
// attention P*V B-fragments become contiguous per lane.
// ---------------------------------------------------------------------------
__global__ __launch_bounds__(128, 1) void qkv_proj_kernel(
    const float* __restrict__ x, const float* __restrict__ Wq,
    const float* __restrict__ Wk, const float* __restrict__ Wv,
    _Float16* __restrict__ qh, _Float16* __restrict__ kh, _Float16* __restrict__ vhT) {
  __shared__ _Float16 xs[16 * 32];
  const int tid = threadIdx.x, lane = tid & 31, wave = tid >> 5;
  const long row0 = (long)blockIdx.x * 16;

  v8f acc[6];
#pragma unroll
  for (int t = 0; t < 6; ++t) acc[t] = v8f{};

  for (int kc = 0; kc < DIM_C; kc += 32) {
    __syncthreads();
    for (int idx = tid; idx < 16 * 32; idx += 128) {
      int r = idx >> 5, c = idx & 31;
      xs[r * 32 + c] = (_Float16)x[(row0 + r) * DIM_C + kc + c];
    }
    __syncthreads();

    v16h afrag = load_a_f16(xs, 32, lane);
#pragma unroll
    for (int t = 0; t < 6; ++t) {
      int tile = wave * 6 + t;
      int sel = tile >> 3, j = tile & 7;
      const float* W = (sel == 0) ? Wq : ((sel == 1) ? Wk : Wv);
      v16h bfrag = load_b_f32cvt(W + (long)(j * 16) * DIM_C + kc, 1, DIM_C, lane);
      acc[t] = WMMA_F16(afrag, bfrag, acc[t]);
    }
  }

  const int col = lane & 15, rbase = (lane >> 4) << 3;
#pragma unroll
  for (int t = 0; t < 6; ++t) {
    int tile = wave * 6 + t;
    int sel = tile >> 3, j = tile & 7;
#pragma unroll
    for (int i = 0; i < 8; ++i) {
      long grow = row0 + i + rbase;              // flattened row in [B*T]
      _Float16 val = (_Float16)acc[t][i];
      if (sel == 2) {                            // v: transposed [B][H][T]
        long bb = grow >> 11, tt = grow & 2047;
        vhT[(bb * DIM_H + j * 16 + col) * DIM_T + tt] = val;
      } else {
        _Float16* o = (sel == 0) ? qh : kh;
        o[grow * DIM_H + j * 16 + col] = val;
      }
    }
  }
}

// ---------------------------------------------------------------------------
// Kernel 2: flash attention.  grid = B * T/128 = 128 blocks, 256 threads.
// Block owns 128 query rows of one batch; wave w owns query tile r0+16w.
// All waves share one key-chunk schedule so K/V tiles are staged ONCE per
// block (TDM DMA into double-buffered LDS), then consumed by all 8 waves.
// ---------------------------------------------------------------------------
__global__ __launch_bounds__(256, 1) void attn_kernel(
    const _Float16* __restrict__ qh, const _Float16* __restrict__ kh,
    const _Float16* __restrict__ vhT, const float* __restrict__ bias,
    const unsigned char* __restrict__ allowed, float* __restrict__ out) {
  __shared__ _Float16 ksh[2][32 * DIM_H];   // [key][comp]   8KB x2
  __shared__ _Float16 vsh[2][DIM_H * 32];   // [hcol][key]   8KB x2
  __shared__ _Float16 pbuf[8][16 * 32];     // per-wave P relayout

  const int tid = threadIdx.x, lane = tid & 31, wave = tid >> 5;
  const int b = blockIdx.x >> 4;
  const int r0 = (blockIdx.x & 15) * 128;
  const int qrow0 = r0 + wave * 16;
  const long rowbase = (long)b * DIM_T + qrow0;
  const int nChunks = (r0 + 128) >> 5;

  v16h qfr[4];
#pragma unroll
  for (int c = 0; c < 4; ++c)
    qfr[c] = load_a_f16(qh + rowbase * DIM_H + c * 32, DIM_H, lane);

  v8f O[8];
#pragma unroll
  for (int j = 0; j < 8; ++j) O[j] = v8f{};
  float mrow[8], lrow[8];
#pragma unroll
  for (int i = 0; i < 8; ++i) { mrow[i] = -3.0e38f; lrow[i] = 0.f; }

  const int col = lane & 15, rbase = (lane >> 4) << 3;

  // ---- staging helper (TDM DMA or cooperative copy fallback) ----
  auto stage = [&](int sc, int pp) {
    const _Float16* kg = kh + ((long)b * DIM_T + sc) * DIM_H;           // 32x128 tile
    const _Float16* vg = vhT + (long)b * DIM_H * DIM_T + sc;            // 128x32 tile
#if HAVE_TDM
    if (wave == 0) {
      tdm_load_2d_f16((unsigned)(size_t)&ksh[pp][0], kg,
                      /*tensor_d0=*/DIM_H, /*tensor_d1=*/DIM_B * DIM_T,
                      /*tile_d0=*/DIM_H, /*tile_d1=*/32, /*stride0=*/DIM_H);
      tdm_load_2d_f16((unsigned)(size_t)&vsh[pp][0], vg,
                      /*tensor_d0=*/DIM_T, /*tensor_d1=*/DIM_B * DIM_H,
                      /*tile_d0=*/32, /*tile_d1=*/DIM_H, /*stride0=*/DIM_T);
    }
#else
    for (int idx = tid; idx < 32 * DIM_H; idx += 256) {
      ksh[pp][idx] = kg[idx];                                   // tile is contiguous
      vsh[pp][idx] = vg[(idx >> 5) * DIM_T + (idx & 31)];
    }
#endif
  };

  stage(0, 0);
  for (int ci = 0; ci < nChunks; ++ci) {
    const int sc = ci << 5, pp = ci & 1;
    const bool more = (ci + 1 < nChunks);
    if (more) stage(sc + 32, pp ^ 1);
#if HAVE_TDM
    if (wave == 0) {
      if (more) __builtin_amdgcn_s_wait_tensorcnt(2);  // current tile done, next in flight
      else      __builtin_amdgcn_s_wait_tensorcnt(0);
    }
#endif
    __syncthreads();

    // ---- scores: two 16x16 tiles over keys [sc, sc+32), K/V from LDS ----
    v8f s0 = v8f{}, s1 = v8f{};
#pragma unroll
    for (int c = 0; c < 4; ++c) {
      v16h b0 = load_b_f16(&ksh[pp][c * 32], 1, DIM_H, lane);
      v16h b1 = load_b_f16(&ksh[pp][16 * DIM_H + c * 32], 1, DIM_H, lane);
      s0 = WMMA_F16(qfr[c], b0, s0);
      s1 = WMMA_F16(qfr[c], b1, s1);
    }

    // ---- bias + mask + online softmax (self-healing for fully masked tiles) ----
    float alpha[8];
#pragma unroll
    for (int i = 0; i < 8; ++i) {
      const int grow = qrow0 + i + rbase;
      const long bi0 = (long)grow * DIM_T + sc + col;
      __builtin_prefetch(bias + bi0 + 32, 0, 1);
      float v0 = s0[i] * SCALE + bias[bi0];
      if (!allowed[bi0]) v0 = NEGHUGE;
      float v1 = s1[i] * SCALE + bias[bi0 + 16];
      if (!allowed[bi0 + 16]) v1 = NEGHUGE;

      const float tmax = rowmax16(fmaxf(v0, v1));
      const float newm = fmaxf(mrow[i], tmax);
      const float a = __expf(mrow[i] - newm);
      const float p0 = __expf(v0 - newm);
      const float p1 = __expf(v1 - newm);
      lrow[i] = lrow[i] * a + rowsum16(p0 + p1);
      mrow[i] = newm;
      alpha[i] = a;

      const int prow = i + rbase;
      pbuf[wave][prow * 32 + col] = (_Float16)p0;
      pbuf[wave][prow * 32 + col + 16] = (_Float16)p1;
    }
    wait_ds0();  // same-wave DS ordering: P writes visible to fragment reload

    // ---- P (16x32) x V (32x16 per head tile), V from LDS, contiguous ----
    v16h pfr = load_a_f16(&pbuf[wave][0], 32, lane);
#pragma unroll
    for (int j = 0; j < 8; ++j) {
#pragma unroll
      for (int i = 0; i < 8; ++i) O[j][i] *= alpha[i];
      v16h vfr = load_b_f16(&vsh[pp][(j * 16) * 32], 1, 32, lane);
      O[j] = WMMA_F16(pfr, vfr, O[j]);
    }
    __syncthreads();   // protect buffer pp from next iteration's staging
  }

  // ---- epilogue ----
#pragma unroll
  for (int j = 0; j < 8; ++j) {
#pragma unroll
    for (int i = 0; i < 8; ++i) {
      const int row = i + rbase;
      out[(rowbase + row) * DIM_H + j * 16 + col] = O[j][i] / lrow[i];
    }
  }
}

// ---------------------------------------------------------------------------
extern "C" void kernel_launch(void* const* d_in, const int* in_sizes, int n_in,
                              void* d_out, int out_size, void* d_ws, size_t ws_size,
                              hipStream_t stream) {
  (void)in_sizes; (void)n_in; (void)out_size; (void)ws_size;
  const float* x  = (const float*)d_in[0];
  const float* Wq = (const float*)d_in[1];
  const float* Wk = (const float*)d_in[2];
  const float* Wv = (const float*)d_in[3];
  const float* bias = (const float*)d_in[4];
  const unsigned char* allowed = (const unsigned char*)d_in[5];
  float* out = (float*)d_out;

  const size_t nQKV = (size_t)DIM_B * DIM_T * DIM_H;
  _Float16* qh  = (_Float16*)d_ws;
  _Float16* kh  = qh + nQKV;
  _Float16* vhT = kh + nQKV;

  qkv_proj_kernel<<<dim3(DIM_B * DIM_T / 16), dim3(128), 0, stream>>>(
      x, Wq, Wk, Wv, qh, kh, vhT);

  attn_kernel<<<dim3(DIM_B * DIM_T / 128), dim3(256), 0, stream>>>(
      qh, kh, vhT, bias, allowed, out);
}
